// CausalTemporalMambaEncoder_28226525069460
// MI455X (gfx1250) — matrix operand order repaired and checked by reading, not compiled, not run on hardware
//
#include <hip/hip_runtime.h>
#include <hip/hip_bf16.h>

// ---------------------------------------------------------------------------
// CausalTemporalMambaEncoder for MI455X (gfx1250, wave32, WMMA + TDM).
// B=4, N=2048 tokens (8192 rows), DM=256, DI=512, DS=16, DTR=16, K=4, L=4.
// GEMMs via v_wmma_f32_16x16x32_bf16; A tiles staged to LDS with the
// Tensor Data Mover (tensor_load_to_lds, TENSORcnt).
// ---------------------------------------------------------------------------

typedef __attribute__((ext_vector_type(16))) __bf16 v16bf;
typedef __attribute__((ext_vector_type(8)))  float v8f;
typedef __attribute__((ext_vector_type(4)))  unsigned v4u;
typedef __attribute__((ext_vector_type(8)))  unsigned v8u;

union Frag16 { unsigned u[8]; v16bf v; };

__device__ __forceinline__ unsigned short f2bf(float f) {
    unsigned u = __float_as_uint(f);
    unsigned r = u + 0x7FFFu + ((u >> 16) & 1u);   // round-to-nearest-even
    return (unsigned short)(r >> 16);
}

__device__ __forceinline__ float fsilu(float x) {
    return x / (1.0f + __expf(-x));
}

__device__ __forceinline__ float fsoftplus(float x) {
    return (x > 20.0f) ? x : log1pf(__expf(x));
}

// ---------------------------------------------------------------------------
// f32 -> bf16 conversion
// ---------------------------------------------------------------------------
__global__ __launch_bounds__(256) void cvt_bf16_kernel(
    const float* __restrict__ in, unsigned short* __restrict__ out, int n)
{
    int i = blockIdx.x * 256 + threadIdx.x;
    if (i < n) out[i] = f2bf(in[i]);
}

// ---------------------------------------------------------------------------
// Token build + first embed matmul (token dim = 3, third component always 0):
// h1 = relu(tok @ We1 + be1), bf16 out. Row = b*2048 + t.
// ---------------------------------------------------------------------------
__global__ __launch_bounds__(256) void embed1_kernel(
    const float* __restrict__ xc, const float* __restrict__ yc,
    const float* __restrict__ xt, const float* __restrict__ yt,
    const float* __restrict__ We1, const float* __restrict__ be1,
    unsigned short* __restrict__ h1bf)
{
    int gid = blockIdx.x * 256 + threadIdx.x;     // 8192*256 total
    int row = gid >> 8;
    int d   = gid & 255;
    int b   = row >> 11;
    int t   = row & 2047;

    float x = (t < 1792) ? xc[b * 1792 + t] : xt[b * 256 + (t - 1792)];
    float yp = 0.0f;
    if (t > 0) {
        int tp = t - 1;
        yp = (tp < 1792) ? yc[b * 1792 + tp] : yt[b * 256 + (tp - 1792)];
    }
    float h = x * We1[d] + yp * We1[256 + d] + be1[d];   // We1 row 2 * 0
    h1bf[(size_t)row * 256 + d] = f2bf(h > 0.0f ? h : 0.0f);
}

// ---------------------------------------------------------------------------
// WMMA GEMM: C[M,N] = A[M,K](bf16,row-major) x B[K,N](bf16,row-major)
// MODE 0: store;  MODE 1: +bias then store;  MODE 2: += existing C (residual).
// Block tile 128(M) x 16(N); 8 waves, one 16x16 WMMA tile each; K step = 64
// (2 WMMAs per barrier). A tile (128x64 bf16) staged into LDS by the Tensor
// Data Mover (wave 0 issues tensor_load_to_lds, waits TENSORcnt). B tile
// staged transposed [n][k] so fragment K-pairs are 16B-contiguous.
// N % 16 == 0, M % 128 == 0, K % 64 == 0 at all call sites.
// ---------------------------------------------------------------------------
template <int MODE>
__global__ __launch_bounds__(256) void gemm_bf16_kernel(
    const unsigned short* __restrict__ A,
    const unsigned short* __restrict__ Bw,
    const float* __restrict__ bias,
    float* __restrict__ C,
    int M, int N, int K)
{
    __shared__ __align__(16) unsigned short At[128 * 64];   // 16 KB (TDM dest)
    __shared__ __align__(16) unsigned short Bt[16 * 64];    //  2 KB, [n][k]

    const int tid  = threadIdx.x;
    const int wave = tid >> 5;
    const int lane = tid & 31;
    const int m16  = lane & 15;
    const int hi   = lane >> 4;
    const int row0 = blockIdx.x * 128;
    const int col0 = blockIdx.y * 16;

    const unsigned ldsA = (unsigned)(size_t)At;   // LDS aperture: low 32 bits

    v8f acc = {};

    for (int k0 = 0; k0 < K; k0 += 64) {
        __syncthreads();   // previous iteration's LDS readers are done

        if (wave == 0) {
            // ---- Tensor DMA descriptor (D#) for the 128x64 bf16 A tile
            unsigned long long ga =
                (unsigned long long)(const void*)A +
                2ull * ((size_t)row0 * (size_t)K + (size_t)k0);
            v4u g0 = { 1u,                                   // count=1, user mode
                       ldsA,                                 // lds_addr
                       (unsigned)ga,                         // global_addr[31:0]
                       ((unsigned)(ga >> 32) & 0x01FFFFFFu)  // global_addr[56:32]
                           | 0x80000000u };                  // type=2 (image)
            v8u g1 = { 0x00010000u,                          // data_size=1 (2B)
                       ((unsigned)K & 0xFFFFu) << 16,        // tensor_dim0 lo
                       (((unsigned)K >> 16) & 0xFFFFu)       // tensor_dim0 hi
                           | (((unsigned)M & 0xFFFFu) << 16),// tensor_dim1 lo
                       (((unsigned)M >> 16) & 0xFFFFu)       // tensor_dim1 hi
                           | (64u << 16),                    // tile_dim0 = 64
                       128u,                                 // tile_dim1 = 128
                       (unsigned)K,                          // dim0_stride lo
                       0u, 0u };                             // dim0_stride hi, dim1_stride
            asm volatile("tensor_load_to_lds %0, %1"
                         :: "s"(g0), "s"(g1) : "memory");
            __builtin_amdgcn_s_wait_tensorcnt(0);
        }

        // ---- stage B tile transposed: Bt[n][k] = Bw[(k0+k)*N + col0+n]
        #pragma unroll
        for (int it = 0; it < 4; ++it) {
            int idx = tid + it * 256;   // 1024 elements
            int k = idx >> 4;
            int n = idx & 15;
            Bt[n * 64 + k] = Bw[(size_t)(k0 + k) * N + col0 + n];
        }
        __syncthreads();

        // ---- gather fragments (2x ds_load_b128 each) + 2 WMMAs
        const uint4* AtV = reinterpret_cast<const uint4*>(At);
        const uint4* BtV = reinterpret_cast<const uint4*>(Bt);
        const int arow = wave * 16 + m16;
        #pragma unroll
        for (int kk = 0; kk < 2; ++kk) {
            Frag16 fa, fb;
            int abase = arow * 64 + kk * 32 + 8 * hi;       // element index
            ((uint4*)fa.u)[0] = AtV[abase >> 3];
            ((uint4*)fa.u)[1] = AtV[(abase + 16) >> 3];
            int bbase = m16 * 64 + kk * 32 + 8 * hi;
            ((uint4*)fb.u)[0] = BtV[bbase >> 3];
            ((uint4*)fb.u)[1] = BtV[(bbase + 16) >> 3];

            acc = __builtin_amdgcn_wmma_f32_16x16x32_bf16(
                /*neg_a=*/false, fa.v, /*neg_b=*/false, fb.v,
                /*c_mod=*/(short)0, acc, /*reuse_a=*/false, /*reuse_b=*/false);
        }
    }

    // ---- epilogue: D layout — lane(n=m16, half=hi), VGPR r -> row r + 8*hi
    const int col = col0 + m16;
    #pragma unroll
    for (int r = 0; r < 8; ++r) {
        int grow = row0 + wave * 16 + r + 8 * hi;
        float v = acc[r];
        if (MODE == 1) v += bias[col];
        size_t o = (size_t)grow * N + col;
        if (MODE == 2) v += C[o];
        C[o] = v;
    }
}

// ---------------------------------------------------------------------------
// RMSNorm over DM=256: xb = bf16(z * rsqrt(mean(z^2)+1e-5) * nw). 1 row/block.
// ---------------------------------------------------------------------------
__global__ __launch_bounds__(256) void rmsnorm_kernel(
    const float* __restrict__ z, const float* __restrict__ nw,
    unsigned short* __restrict__ xb)
{
    __shared__ float red[256];
    const int row = blockIdx.x;
    const int d   = threadIdx.x;
    float v = z[(size_t)row * 256 + d];
    red[d] = v * v;
    __syncthreads();
    #pragma unroll
    for (int s = 128; s > 0; s >>= 1) {
        if (d < s) red[d] += red[d + s];
        __syncthreads();
    }
    float rs = rsqrtf(red[0] * (1.0f / 256.0f) + 1e-5f);
    xb[(size_t)row * 256 + d] = f2bf(v * rs * nw[d]);
}

// ---------------------------------------------------------------------------
// Depthwise causal conv (K=4) + bias + silu on u = ug[:, 0:512].
// Writes f32 uconv and bf16 copy for the next GEMM.
// ---------------------------------------------------------------------------
__global__ __launch_bounds__(256) void conv_silu_kernel(
    const float* __restrict__ ug, const float* __restrict__ Wc,
    const float* __restrict__ bc,
    float* __restrict__ uconv, unsigned short* __restrict__ ucbf)
{
    int gid = blockIdx.x * 256 + threadIdx.x;     // 8192*512 total
    int row = gid >> 9;
    int d   = gid & 511;
    int b   = row >> 11;
    int t   = row & 2047;

    float s = bc[d];
    #pragma unroll
    for (int j = 0; j < 4; ++j) {
        int tt = t - 3 + j;
        if (tt >= 0)
            s += ug[((size_t)(b * 2048 + tt)) * 1024 + d] * Wc[d * 4 + j];
    }
    s = fsilu(s);
    uconv[(size_t)row * 512 + d] = s;
    ucbf[(size_t)row * 512 + d]  = f2bf(s);
}

// ---------------------------------------------------------------------------
// delta = softplus(dt @ Wdt + bdt); dt = dbl[:, 0:16], Wdt is [16,512].
// ---------------------------------------------------------------------------
__global__ __launch_bounds__(256) void delta_kernel(
    const float* __restrict__ dbl, const float* __restrict__ Wdt,
    const float* __restrict__ bdt, float* __restrict__ delta)
{
    int gid = blockIdx.x * 256 + threadIdx.x;     // 8192*512 total
    int row = gid >> 9;
    int d   = gid & 511;
    const float* dt = dbl + (size_t)row * 48;
    float acc = bdt[d];
    #pragma unroll
    for (int k = 0; k < 16; ++k)
        acc += dt[k] * Wdt[k * 512 + d];
    delta[(size_t)row * 512 + d] = fsoftplus(acc);
}

// ---------------------------------------------------------------------------
// Selective scan: one thread per (b, d) channel, 16 states in registers,
// serial over 2048 timesteps. B_t = dbl[:,16:32], C_t = dbl[:,32:48].
// ---------------------------------------------------------------------------
__global__ __launch_bounds__(256) void scan_kernel(
    const float* __restrict__ delta, const float* __restrict__ uconv,
    const float* __restrict__ dbl,   const float* __restrict__ Alog,
    float* __restrict__ ys)
{
    int gid = blockIdx.x * 256 + threadIdx.x;     // 2048 threads
    int b = gid >> 9;
    int d = gid & 511;

    float A[16];
    #pragma unroll
    for (int s = 0; s < 16; ++s)
        A[s] = -__expf(Alog[d * 16 + s]);

    float h[16];
    #pragma unroll
    for (int s = 0; s < 16; ++s) h[s] = 0.0f;

    for (int t = 0; t < 2048; ++t) {
        size_t row = (size_t)b * 2048 + t;
        float dl = delta[row * 512 + d];
        float uu = uconv[row * 512 + d];
        float du = dl * uu;
        const float* bcv = dbl + row * 48;
        float acc = 0.0f;
        #pragma unroll
        for (int s = 0; s < 16; ++s) {
            h[s] = __expf(dl * A[s]) * h[s] + du * bcv[16 + s];
            acc += h[s] * bcv[32 + s];
        }
        ys[row * 512 + d] = acc;
    }
}

// ---------------------------------------------------------------------------
// y = (ys + Dp*u) * silu(gate); gate = ug[:, 512:1024]. bf16 out for GEMM.
// ---------------------------------------------------------------------------
__global__ __launch_bounds__(256) void mix_kernel(
    const float* __restrict__ ys, const float* __restrict__ uconv,
    const float* __restrict__ ug, const float* __restrict__ Dp,
    unsigned short* __restrict__ ybf)
{
    int gid = blockIdx.x * 256 + threadIdx.x;     // 8192*512 total
    int row = gid >> 9;
    int d   = gid & 511;
    float y = ys[(size_t)row * 512 + d] + Dp[d] * uconv[(size_t)row * 512 + d];
    float g = ug[(size_t)row * 1024 + 512 + d];
    ybf[(size_t)row * 512 + d] = f2bf(y * fsilu(g));
}

// ---------------------------------------------------------------------------
// Host-side orchestration
// ---------------------------------------------------------------------------
extern "C" void kernel_launch(void* const* d_in, const int* in_sizes, int n_in,
                              void* d_out, int out_size, void* d_ws, size_t ws_size,
                              hipStream_t stream)
{
    (void)in_sizes; (void)n_in; (void)out_size; (void)ws_size;

    const float* xc     = (const float*)d_in[0];
    const float* yc     = (const float*)d_in[1];
    const float* xt     = (const float*)d_in[2];
    const float* yt     = (const float*)d_in[3];
    const float* We1    = (const float*)d_in[4];
    const float* be1    = (const float*)d_in[5];
    const float* We2    = (const float*)d_in[6];
    const float* be2    = (const float*)d_in[7];
    const float* norm_w = (const float*)d_in[8];
    const float* W_in   = (const float*)d_in[9];
    const float* W_conv = (const float*)d_in[10];
    const float* b_conv = (const float*)d_in[11];
    const float* W_xprj = (const float*)d_in[12];
    const float* W_dt   = (const float*)d_in[13];
    const float* b_dt   = (const float*)d_in[14];
    const float* A_log  = (const float*)d_in[15];
    const float* Dp     = (const float*)d_in[16];
    const float* W_out  = (const float*)d_in[17];

    float* zout = (float*)d_out;                  // (4,2048,256) f32

    // ---- workspace partition (256B aligned)
    char* w = (char*)d_ws;
    auto alloc = [&](size_t bytes) -> void* {
        void* p = (void*)w;
        w += (bytes + 255) & ~(size_t)255;
        return p;
    };
    const size_t R = 8192;  // rows = B*N
    unsigned short* xb    = (unsigned short*)alloc(R * 256 * 2);   // rmsnorm / h1 bf16
    float*          ug    = (float*)         alloc(R * 1024 * 4);  // x@Wi (u|gate)
    float*          uconv = (float*)         alloc(R * 512 * 4);
    unsigned short* ucbf  = (unsigned short*)alloc(R * 512 * 2);
    float*          dbl   = (float*)         alloc(R * 48 * 4);    // dt|B|C
    float*          dlt   = (float*)         alloc(R * 512 * 4);
    float*          ys    = (float*)         alloc(R * 512 * 4);
    unsigned short* ybf   = (unsigned short*)alloc(R * 512 * 2);
    unsigned short* We2b  = (unsigned short*)alloc(256 * 256 * 2);
    unsigned short* Wib   = (unsigned short*)alloc(4 * 256 * 1024 * 2);
    unsigned short* Wxb   = (unsigned short*)alloc(4 * 512 * 48 * 2);
    unsigned short* Wob   = (unsigned short*)alloc(4 * 512 * 256 * 2);

    // ---- weight conversion to bf16
    auto cvt = [&](const float* src, unsigned short* dst, int n) {
        cvt_bf16_kernel<<<(n + 255) / 256, 256, 0, stream>>>(src, dst, n);
    };
    cvt(We2,    We2b, 256 * 256);
    cvt(W_in,   Wib,  4 * 256 * 1024);
    cvt(W_xprj, Wxb,  4 * 512 * 48);
    cvt(W_out,  Wob,  4 * 512 * 256);

    // ---- embedding: h1 = relu(tok@We1+be1); z = h1 @ We2 + be2
    embed1_kernel<<<8192, 256, 0, stream>>>(xc, yc, xt, yt, We1, be1, xb);
    gemm_bf16_kernel<1><<<dim3(64, 16), 256, 0, stream>>>(
        xb, We2b, be2, zout, 8192, 256, 256);

    // ---- L = 4 mamba blocks
    for (int l = 0; l < 4; ++l) {
        rmsnorm_kernel<<<8192, 256, 0, stream>>>(zout, norm_w + l * 256, xb);

        gemm_bf16_kernel<0><<<dim3(64, 64), 256, 0, stream>>>(
            xb, Wib + (size_t)l * 256 * 1024, nullptr, ug, 8192, 1024, 256);

        conv_silu_kernel<<<16384, 256, 0, stream>>>(
            ug, W_conv + l * 512 * 4, b_conv + l * 512, uconv, ucbf);

        gemm_bf16_kernel<0><<<dim3(64, 3), 256, 0, stream>>>(
            ucbf, Wxb + (size_t)l * 512 * 48, nullptr, dbl, 8192, 48, 512);

        delta_kernel<<<16384, 256, 0, stream>>>(
            dbl, W_dt + l * 16 * 512, b_dt + l * 512, dlt);

        scan_kernel<<<8, 256, 0, stream>>>(
            dlt, uconv, dbl, A_log + l * 512 * 16, ys);

        mix_kernel<<<16384, 256, 0, stream>>>(ys, uconv, ug, Dp + l * 512, ybf);

        gemm_bf16_kernel<2><<<dim3(64, 16), 256, 0, stream>>>(
            ybf, Wob + (size_t)l * 512 * 256, nullptr, zout, 8192, 256, 512);
    }
}
